// MSARCModule_80616536146457
// MI455X (gfx1250) — compile-verified, matrix operand order checked by reading, not run on hardware
//
#include <hip/hip_runtime.h>

typedef __bf16 bf16;
typedef __attribute__((ext_vector_type(8)))  bf16  v8bf;
typedef __attribute__((ext_vector_type(16))) bf16  v16bf;
typedef __attribute__((ext_vector_type(8)))  float v8f;

#define HW    128
#define CPB   128      // channels per branch
#define CTOT  384
#define NBATCH 8
#define NK    4        // kernel_number
#define KDIM  1152     // CPB*9
#define LDP   40       // LDS row pitch in bf16 elements (80 B -> conflict-free b128 frags)

__device__ __forceinline__ float silu_f(float t) { return t / (1.0f + __expf(-t)); }

// CDNA5 async global->LDS copy (16 B per lane), tracked by ASYNCcnt.
__device__ __forceinline__ void async_b128(unsigned lds_byte_off, const void* gsrc) {
    asm volatile("global_load_async_to_lds_b128 %0, %1, off"
                 :: "v"(lds_byte_off), "v"(gsrc) : "memory");
}
__device__ __forceinline__ void wait_async0() {
    asm volatile("s_wait_asynccnt 0x0" ::: "memory");
}
__device__ __forceinline__ unsigned lds_off_of(const void* p) {
    return (unsigned)(uintptr_t)p;   // flat LDS addr low 32 bits = LDS byte address
}

__device__ __forceinline__ v8f wmma_bf16(v16bf a, v16bf b, v8f c) {
    return __builtin_amdgcn_wmma_f32_16x16x32_bf16(false, a, false, b, (short)0, c, false, false);
}

// lanes 0-15: K0-7 + K16-23, lanes 16-31: K8-15 + K24-31 -> two ds_load_b128.
__device__ __forceinline__ v16bf frag_from_lds(const bf16* rowbase) {
    v8bf lo = *(const v8bf*)rowbase;
    v8bf hi = *(const v8bf*)(rowbase + 16);
    return __builtin_shufflevector(lo, hi, 0,1,2,3,4,5,6,7,8,9,10,11,12,13,14,15);
}

// ---------------------------------------------------------------- zero init
__global__ void k_zero(float* __restrict__ p, int n) {
    int i = blockIdx.x * 256 + threadIdx.x;
    if (i < n) p[i] = 0.0f;
}

// ------------------------------------------- routing: dwconv + LN + ReLU + GAP
__global__ void k_routing(const float* __restrict__ x, const float* __restrict__ dwc,
                          const float* __restrict__ lng, const float* __restrict__ lnb,
                          float* __restrict__ pooled) {
    extern __shared__ float tval[];      // [128 ch][128 w]
    int bx = blockIdx.x;
    int h = bx & 127; bx >>= 7;
    int b = bx & 7;   bx >>= 3;
    int i = bx;
    int w = threadIdx.x;

    const float* xb = x + ((size_t)(b * CTOT + i * CPB)) * HW * HW;
    const float* wt = dwc + (size_t)i * CPB * 9;

    float s1 = 0.0f, s2 = 0.0f;
    for (int ch = 0; ch < CPB; ++ch) {
        const float* xc = xb + (size_t)ch * HW * HW;
        const float* wc = wt + ch * 9;
        float t = 0.0f;
#pragma unroll
        for (int ky = 0; ky < 3; ++ky) {
            int hh = h + ky - 1;
            if ((unsigned)hh < HW) {
                const float* row = xc + hh * HW;
#pragma unroll
                for (int kx = 0; kx < 3; ++kx) {
                    int ww = w + kx - 1;
                    float v = ((unsigned)ww < HW) ? row[ww] : 0.0f;
                    t += v * wc[ky * 3 + kx];
                }
            }
        }
        tval[ch * HW + w] = t;
        s1 += t; s2 += t * t;
    }
    float mu   = s1 * (1.0f / CPB);
    float var  = s2 * (1.0f / CPB) - mu * mu;
    float rstd = rsqrtf(var + 1e-5f);

    float* pool = pooled + (i * NBATCH + b) * CPB;
    for (int ch = 0; ch < CPB; ++ch) {
        float y = (tval[ch * HW + w] - mu) * rstd * lng[i * CPB + ch] + lnb[i * CPB + ch];
        y = fmaxf(y, 0.0f);
#pragma unroll
        for (int off = 16; off > 0; off >>= 1) y += __shfl_xor(y, off, 32);
        if ((threadIdx.x & 31) == 0) atomicAdd(&pool[ch], y);
    }
}

// ------------------------------------------- fc heads + rotation matrices
__global__ void k_heads(const float* __restrict__ pooled,
                        const float* __restrict__ faw, const float* __restrict__ fab,
                        const float* __restrict__ ftw, float* __restrict__ Mbuf) {
    __shared__ float p[CPB];
    __shared__ float alphas[NK], thetas[NK];
    int ib = blockIdx.x;
    int i  = ib / NBATCH;
    int t  = threadIdx.x;
    p[t] = pooled[ib * CPB + t] * (1.0f / (HW * HW));
    __syncthreads();
    if (t < 8) {
        int n = t & 3;
        const float* W = (t < 4 ? faw : ftw) + (i * NK + n) * CPB;
        float d = 0.0f;
        for (int c = 0; c < CPB; ++c) d += p[c] * W[c];
        if (t < 4) alphas[n] = 1.0f / (1.0f + __expf(-(d + fab[i * NK + n])));
        else       thetas[n] = d / (1.0f + fabsf(d)) * 0.69813170079773183f;  // 40/180*pi
    }
    __syncthreads();
    for (int idx = t; idx < NK * 81; idx += 128) {
        int n = idx / 81, rest = idx % 81, pp = rest / 9, q = rest % 9;
        float th = thetas[n], cth = __cosf(th), sth = __sinf(th);
        float pxp = (float)(pp % 3 - 1), pyp = (float)(pp / 3 - 1);
        float pxq = (float)(q  % 3 - 1), pyq = (float)(q  / 3 - 1);
        float xr = fminf(fmaxf(pxp * cth - pyp * sth, -1.0f), 1.0f);
        float yr = fminf(fmaxf(pxp * sth + pyp * cth, -1.0f), 1.0f);
        float wx = fmaxf(0.0f, 1.0f - fabsf(xr - pxq));
        float wy = fmaxf(0.0f, 1.0f - fabsf(yr - pyq));
        Mbuf[ib * NK * 81 + idx] = alphas[n] * wx * wy;
    }
}

// ------------------------------------------- per-sample weight synthesis -> bf16
__global__ void k_wsynth(const float* __restrict__ Mbuf, const float* __restrict__ arcw,
                         bf16* __restrict__ Wbuf) {
    __shared__ float Msh[NK * 81];
    int bx = blockIdx.x;
    int ib = bx >> 6, chunk = bx & 63;
    int i  = ib >> 3;
    int t  = threadIdx.x;
    for (int idx = t; idx < NK * 81; idx += 256) Msh[idx] = Mbuf[ib * NK * 81 + idx];
    __syncthreads();
    int pair = chunk * 256 + t;
    int co = pair >> 7, ci = pair & 127;
    float acc[9];
#pragma unroll
    for (int p = 0; p < 9; ++p) acc[p] = 0.0f;
    for (int n = 0; n < NK; ++n) {
        const float* wsrc = arcw + ((((size_t)i * NK + n) * CPB + co) * CPB + ci) * 9;
        float wq[9];
#pragma unroll
        for (int q = 0; q < 9; ++q) wq[q] = wsrc[q];
        const float* Mn = &Msh[n * 81];
#pragma unroll
        for (int p = 0; p < 9; ++p) {
            float a = 0.0f;
#pragma unroll
            for (int q = 0; q < 9; ++q) a += Mn[p * 9 + q] * wq[q];
            acc[p] += a;
        }
    }
    bf16* dst = Wbuf + ((size_t)ib * CPB + co) * KDIM + ci;
#pragma unroll
    for (int p = 0; p < 9; ++p) dst[p * CPB] = (bf16)acc[p];
}

// ------------------------------------------- f32 -> bf16 convert (fusion weights)
__global__ void k_cvt(const float* __restrict__ src, bf16* __restrict__ dst, int n) {
    int i = blockIdx.x * 256 + threadIdx.x;
    if (i < n) dst[i] = (bf16)src[i];
}

// ------------------------------------------- branch implicit-GEMM (WMMA) + BN1 + SiLU
// grid: 3*8*128 blocks (i,b,row); 256 threads = 8 waves; block tile 128Co x 128pix.
// Each wave: 32Co x 64pix -> 8 WMMAs per 32-wide K-step. A tile staged via async LDS DMA.
__global__ void __launch_bounds__(256) k_branchgemm(
        const float* __restrict__ x, const bf16* __restrict__ Wbuf,
        const float* __restrict__ g1, const float* __restrict__ b1,
        const float* __restrict__ m1, const float* __restrict__ v1,
        bf16* __restrict__ fbuf) {
    __shared__ bf16 As[CPB * LDP];   // 128 co x 32 k
    __shared__ bf16 Bs[HW * LDP];    // 128 n  x 32 k
    int bx = blockIdx.x;
    int h = bx & 127; bx >>= 7;
    int b = bx & 7;   bx >>= 3;
    int i = bx;
    int ib  = i * NBATCH + b;
    int dil = i + 1;
    int t = threadIdx.x, lane = t & 31, wv = t >> 5;
    int co0 = (wv & 3) << 5;     // wave Co band: 0/32/64/96
    int n0  = (wv >> 2) << 6;    // wave pixel band: 0/64

    v8f acc[2][4];
#pragma unroll
    for (int m = 0; m < 2; ++m)
#pragma unroll
        for (int s = 0; s < 4; ++s)
#pragma unroll
            for (int r = 0; r < 8; ++r) acc[m][s][r] = 0.0f;

    const bf16*  Wb = Wbuf + (size_t)ib * CPB * KDIM;
    const float* xb = x + ((size_t)b * CTOT + i * CPB) * HW * HW;

    int aco = t >> 1, akh = (t & 1) << 4;              // A staging role
    unsigned alds = lds_off_of(&As[aco * LDP + akh]);
    int bn = t & 127, bc0 = (t >> 7) << 4;             // B staging role: 16 ci per thread
    int m15 = lane & 15;
    int koff = (lane < 16) ? 0 : 8;

    for (int tap = 0; tap < 9; ++tap) {
        int dy = tap / 3 - 1, dx = tap % 3 - 1;
        int srow = h + dy * dil;
        int scol = bn + dx * dil;
        bool ok = ((unsigned)srow < HW) && ((unsigned)scol < HW);
        int srow_c = min(max(srow, 0), HW - 1);
        int scol_c = min(max(scol, 0), HW - 1);
        for (int cc = 0; cc < 4; ++cc) {
            // A tile: async global->LDS (bf16 copy, no VGPR round-trip)
            const bf16* asrc = Wb + (size_t)aco * KDIM + tap * CPB + (cc << 5) + akh;
            async_b128(alds,      asrc);
            async_b128(alds + 16, asrc + 8);
            // B tile: f32 load (clamped addr + select, branch-free) -> bf16 LDS
            const float* xcol = xb + ((size_t)((cc << 5) + bc0) * HW + srow_c) * HW + scol_c;
#pragma unroll
            for (int r = 0; r < 16; ++r) {
                float v = xcol[(size_t)r * HW * HW];
                v = ok ? v : 0.0f;
                Bs[bn * LDP + bc0 + r] = (bf16)v;
            }
            wait_async0();
            __syncthreads();
            v16bf a0 = frag_from_lds(&As[(co0 +      m15) * LDP + koff]);
            v16bf a1 = frag_from_lds(&As[(co0 + 16 + m15) * LDP + koff]);
#pragma unroll
            for (int s = 0; s < 4; ++s) {
                v16bf bfr = frag_from_lds(&Bs[(n0 + (s << 4) + m15) * LDP + koff]);
                acc[0][s] = wmma_bf16(a0, bfr, acc[0][s]);
                acc[1][s] = wmma_bf16(a1, bfr, acc[1][s]);
            }
            __syncthreads();
        }
    }
    // epilogue: BN1 + SiLU, bf16 store into channel-last concat buffer (L2-resident)
    int rsel = (lane < 16) ? 0 : 8;
    int ncol = lane & 15;
    float sc[2][8], sh[2][8];
#pragma unroll
    for (int m = 0; m < 2; ++m)
#pragma unroll
        for (int r = 0; r < 8; ++r) {
            int cg = i * CPB + co0 + (m << 4) + r + rsel;
            float scl = g1[cg] * rsqrtf(v1[cg] + 1e-3f);
            sc[m][r] = scl; sh[m][r] = b1[cg] - m1[cg] * scl;
        }
    bf16* fb = fbuf + ((size_t)(b * HW + h) * HW) * CTOT + i * CPB;
#pragma unroll
    for (int m = 0; m < 2; ++m)
#pragma unroll
        for (int s = 0; s < 4; ++s)
#pragma unroll
            for (int r = 0; r < 8; ++r) {
                float y = silu_f(acc[m][s][r] * sc[m][r] + sh[m][r]);
                int w  = n0 + (s << 4) + ncol;
                int co = co0 + (m << 4) + r + rsel;
                fb[(size_t)w * CTOT + co] = (bf16)y;
            }
}

// ------------------------------------------- 1x1 fusion GEMM (WMMA) + BN2 + SiLU + res
// grid: 8*128*3 blocks; 256 threads; tile 128Co x 128pix; A and B both async-staged.
__global__ void __launch_bounds__(256) k_fusion(
        const bf16* __restrict__ fbuf, const bf16* __restrict__ fw,
        const float* __restrict__ fbias,
        const float* __restrict__ g2, const float* __restrict__ b2,
        const float* __restrict__ m2, const float* __restrict__ v2,
        const float* __restrict__ x, float* __restrict__ out) {
    __shared__ bf16 As[CPB * LDP];
    __shared__ bf16 Bs[HW * LDP];
    int bx = blockIdx.x;
    int band = bx % 3; bx /= 3;
    int h = bx & 127;  bx >>= 7;
    int b = bx;
    int t = threadIdx.x, lane = t & 31, wv = t >> 5;
    int co0 = (wv & 3) << 5;
    int n0  = (wv >> 2) << 6;

    v8f acc[2][4];
#pragma unroll
    for (int m = 0; m < 2; ++m)
#pragma unroll
        for (int s = 0; s < 4; ++s)
#pragma unroll
            for (int r = 0; r < 8; ++r) acc[m][s][r] = 0.0f;

    int arow = t >> 1, akh = (t & 1) << 4;   // row role for both A (co) and B (pixel)
    unsigned alds = lds_off_of(&As[arow * LDP + akh]);
    unsigned blds = lds_off_of(&Bs[arow * LDP + akh]);
    const bf16* fwrow = fw + (size_t)(band * CPB + arow) * CTOT + akh;
    const bf16* fbrow = fbuf + ((size_t)(b * HW + h) * HW + arow) * CTOT + akh;  // channel-last
    int m15 = lane & 15;
    int koff = (lane < 16) ? 0 : 8;

    for (int kk = 0; kk < 12; ++kk) {
        int kb = kk << 5;
        async_b128(alds,      fwrow + kb);
        async_b128(alds + 16, fwrow + kb + 8);
        async_b128(blds,      fbrow + kb);
        async_b128(blds + 16, fbrow + kb + 8);
        wait_async0();
        __syncthreads();
        v16bf a0 = frag_from_lds(&As[(co0 +      m15) * LDP + koff]);
        v16bf a1 = frag_from_lds(&As[(co0 + 16 + m15) * LDP + koff]);
#pragma unroll
        for (int s = 0; s < 4; ++s) {
            v16bf bfr = frag_from_lds(&Bs[(n0 + (s << 4) + m15) * LDP + koff]);
            acc[0][s] = wmma_bf16(a0, bfr, acc[0][s]);
            acc[1][s] = wmma_bf16(a1, bfr, acc[1][s]);
        }
        __syncthreads();
    }
    int rsel = (lane < 16) ? 0 : 8;
    int ncol = lane & 15;
    float sc[2][8], sh[2][8], bi[2][8];
#pragma unroll
    for (int m = 0; m < 2; ++m)
#pragma unroll
        for (int r = 0; r < 8; ++r) {
            int o = band * CPB + co0 + (m << 4) + r + rsel;
            float scl = g2[o] * rsqrtf(v2[o] + 1e-3f);
            sc[m][r] = scl; sh[m][r] = b2[o] - m2[o] * scl; bi[m][r] = fbias[o];
        }
#pragma unroll
    for (int m = 0; m < 2; ++m)
#pragma unroll
        for (int s = 0; s < 4; ++s)
#pragma unroll
            for (int r = 0; r < 8; ++r) {
                int o = band * CPB + co0 + (m << 4) + r + rsel;
                int w = n0 + (s << 4) + ncol;
                float y = silu_f((acc[m][s][r] + bi[m][r]) * sc[m][r] + sh[m][r]);
                size_t idx = (((size_t)b * CTOT + o) * HW + h) * HW + w;
                out[idx] = y + x[idx];
            }
}

// ---------------------------------------------------------------- launcher
extern "C" void kernel_launch(void* const* d_in, const int* in_sizes, int n_in,
                              void* d_out, int out_size, void* d_ws, size_t ws_size,
                              hipStream_t stream) {
    const float* x    = (const float*)d_in[0];
    const float* dwcw = (const float*)d_in[1];
    const float* lng  = (const float*)d_in[2];
    const float* lnb  = (const float*)d_in[3];
    const float* faw  = (const float*)d_in[4];
    const float* fab  = (const float*)d_in[5];
    const float* ftw  = (const float*)d_in[6];
    const float* arcw = (const float*)d_in[7];
    const float* bn1g = (const float*)d_in[8];
    const float* bn1b = (const float*)d_in[9];
    const float* bn1m = (const float*)d_in[10];
    const float* bn1v = (const float*)d_in[11];
    const float* fusw = (const float*)d_in[12];
    const float* fusb = (const float*)d_in[13];
    const float* bn2g = (const float*)d_in[14];
    const float* bn2b = (const float*)d_in[15];
    const float* bn2m = (const float*)d_in[16];
    const float* bn2v = (const float*)d_in[17];
    float* out = (float*)d_out;

    char* ws = (char*)d_ws;
    size_t off = 0;
    auto carve = [&](size_t bytes) { size_t o = off; off += (bytes + 255) & ~(size_t)255; return o; };
    float* pooled = (float*)(ws + carve((size_t)3 * 8 * 128 * 4));
    float* Mbuf   = (float*)(ws + carve((size_t)3 * 8 * 4 * 81 * 4));
    bf16*  fuswb  = (bf16*) (ws + carve((size_t)384 * 384 * 2));
    bf16*  Wbuf   = (bf16*) (ws + carve((size_t)24 * 128 * KDIM * 2));
    bf16*  fbuf   = (bf16*) (ws + carve((size_t)8 * 128 * 128 * 384 * 2));  // channel-last
    (void)ws_size; (void)in_sizes; (void)n_in; (void)out_size;

    k_zero<<<(3 * 8 * 128 + 255) / 256, 256, 0, stream>>>(pooled, 3 * 8 * 128);
    k_routing<<<3 * 8 * 128, 128, 128 * 128 * sizeof(float), stream>>>(x, dwcw, lng, lnb, pooled);
    k_heads<<<24, 128, 0, stream>>>(pooled, faw, fab, ftw, Mbuf);
    k_cvt<<<(384 * 384 + 255) / 256, 256, 0, stream>>>(fusw, fuswb, 384 * 384);
    k_wsynth<<<24 * 64, 256, 0, stream>>>(Mbuf, arcw, Wbuf);
    k_branchgemm<<<3 * 8 * 128, 256, 0, stream>>>(x, Wbuf, bn1g, bn1b, bn1m, bn1v, fbuf);
    k_fusion<<<8 * 128 * 3, 256, 0, stream>>>(fbuf, fuswb, fusb, bn2g, bn2b, bn2m, bn2v, x, out);
}
